// DeepseekMoE_29635274342473
// MI455X (gfx1250) — compile-verified, matrix operand order checked by reading, not run on hardware
//
#include <hip/hip_runtime.h>

// ---------------------------------------------------------------------------
// DeepSeek MoE on MI455X (gfx1250): bf16 WMMA (v_wmma_f32_16x16x32_bf16),
// gather-based routed experts (TOP_K=2 of 8), fused SwiGLU, deterministic
// combine. K-loop data movement:
//   - B (weight) tiles: Tensor Data Mover (tensor_load_to_lds, TENSORcnt),
//     one descriptor per 128x32 tile issued by wave 0.
//   - A (activation) tiles: per-lane async global->LDS loads (ASYNCcnt),
//     needed because routed-expert rows are gathered.
// Double-buffered LDS, one barrier per K-step.
// ---------------------------------------------------------------------------

#define T_TOK 8192
#define D_DIM 2048
#define E_EXP 8
#define H_DIM 1408
#define SH_DIM 2816
#define KT 32

typedef __attribute__((ext_vector_type(16))) __bf16 v16bf;
typedef __attribute__((ext_vector_type(8)))  float  v8f;
typedef __attribute__((ext_vector_type(4)))  unsigned int u32x4;
typedef __attribute__((ext_vector_type(8)))  int          i32x8;
typedef __attribute__((ext_vector_type(4)))  int          i32x4;

union FragU { v16bf v; uint4 q[2]; };

__device__ __forceinline__ unsigned short f2bf(float f) {
  unsigned int u = __builtin_bit_cast(unsigned int, f);
  unsigned int r = u + 0x7FFFu + ((u >> 16) & 1u);
  return (unsigned short)(r >> 16);
}

// Async global -> LDS (16B per lane), tracked by ASYNCcnt. Generic pointers to
// __shared__ carry the wave-relative LDS byte offset in their low 32 bits.
__device__ __forceinline__ void async_ld_b128(void* lds_ptr, const void* gptr) {
  unsigned int l = (unsigned int)(unsigned long long)lds_ptr;
  unsigned long long g = (unsigned long long)gptr;
  asm volatile("global_load_async_to_lds_b128 %0, %1, off"
               :: "v"(l), "v"(g) : "memory");
}
__device__ __forceinline__ void wait_async0() {
  asm volatile("s_wait_asynccnt 0x0" ::: "memory");
}

// TDM: DMA one 2D tile (tile_rows x tile_k elements, 2 bytes each) of a
// row-major [Nrows, K] bf16 tensor into LDS. D# per CDNA5 ISA 8.3-8.5.
// This toolchain (clang-23 / therock-10.0) exposes the 6-arg builtin:
// (u32x4 g0, i32x8 g1, i32x4 g2, i32x4 g3, i32x8 extra, i32 cpol);
// groups 2/3/extra are only needed for 3D+/gather descriptors -> zeros.
__device__ __forceinline__ void tdm_load_2d(void* lds, const unsigned short* g,
                                            int K, int Nrows, int tile_rows,
                                            int tile_k) {
  unsigned long long ga = (unsigned long long)g;
  u32x4 g0;
  g0[0] = 1u;                                        // count=1, user mode
  g0[1] = (unsigned int)(unsigned long long)lds;     // lds_addr (bytes)
  g0[2] = (unsigned int)ga;                          // global_addr[31:0]
  g0[3] = (unsigned int)((ga >> 32) & 0x01FFFFFFu) | (2u << 30); // ga[56:32]|type=2
  i32x8 g1;
  g1[0] = (int)(1u << 16);                           // data_size=1 (2 bytes)
  g1[1] = (int)(((unsigned)K & 0xFFFFu) << 16);      // tensor_dim0[15:0]
  g1[2] = (int)(((unsigned)K >> 16) |
                (((unsigned)Nrows & 0xFFFFu) << 16)); // dim0[31:16] | dim1[15:0]
  g1[3] = (int)(((unsigned)Nrows >> 16) |
                ((unsigned)tile_k << 16));            // dim1[31:16] | tile_dim0
  g1[4] = tile_rows;                                  // tile_dim1 (tile_dim2=0)
  g1[5] = K;                                          // tensor_dim0_stride[31:0]
  g1[6] = 0;
  g1[7] = 0;
  i32x4 z4 = {0, 0, 0, 0};
  i32x8 z8 = {0, 0, 0, 0, 0, 0, 0, 0};
  __builtin_amdgcn_tensor_load_to_lds(g0, g1, z4, z4, z8, 0);
}
__device__ __forceinline__ void wait_tensor0() {
  __builtin_amdgcn_s_wait_tensorcnt(0);
}

// --------------------------- fp32 -> bf16 convert ---------------------------
__global__ void cvt_f32_bf16(const float* __restrict__ s,
                             unsigned short* __restrict__ d, size_t n) {
  size_t i = (size_t)blockIdx.x * blockDim.x + threadIdx.x;
  size_t st = (size_t)gridDim.x * blockDim.x;
  for (; i < n; i += st) d[i] = f2bf(s[i]);
}

// ------------------------------- gating -------------------------------------
__global__ void __launch_bounds__(256)
gate_kernel(const float* __restrict__ x, const float* __restrict__ gw,
            int* __restrict__ sel, float* __restrict__ rw,
            int* __restrict__ counts, int* __restrict__ list,
            int* __restrict__ posmap) {
  int lane = threadIdx.x & 31;
  int wid  = threadIdx.x >> 5;
  int t = blockIdx.x * 8 + wid;
  const float* xr = x + (size_t)t * D_DIM;
  float lg[E_EXP];
  for (int e = 0; e < E_EXP; ++e) {
    const float* gr = gw + (size_t)e * D_DIM;
    float s = 0.f;
    for (int d = lane; d < D_DIM; d += 32) s += xr[d] * gr[d];
    for (int o = 16; o > 0; o >>= 1) s += __shfl_xor(s, o, 32);
    lg[e] = s;
  }
  if (lane == 0) {
    int i0 = 0; float v0 = lg[0];
    for (int e = 1; e < E_EXP; ++e) if (lg[e] > v0) { v0 = lg[e]; i0 = e; }
    int i1 = (i0 == 0) ? 1 : 0; float v1 = lg[i1];
    for (int e = 0; e < E_EXP; ++e)
      if (e != i0 && lg[e] > v1) { v1 = lg[e]; i1 = e; }
    float w1v = expf(v1 - v0);
    float p0 = 1.f / (1.f + w1v);
    float p1 = w1v * p0;
    sel[2 * t] = i0; sel[2 * t + 1] = i1;
    rw[2 * t] = p0;  rw[2 * t + 1] = p1;
    int p = atomicAdd(&counts[i0], 1);
    list[i0 * T_TOK + p] = t; posmap[2 * t] = p;
    int q = atomicAdd(&counts[i1], 1);
    list[i1 * T_TOK + q] = t; posmap[2 * t + 1] = q;
  }
}

// Deterministic s_e + exclusive prefix offsets. One block, 8 waves.
__global__ void __launch_bounds__(256)
se_off_kernel(const int* __restrict__ sel, const float* __restrict__ rw,
              const int* __restrict__ counts, float* __restrict__ s_e,
              int* __restrict__ offs) {
  int lane = threadIdx.x & 31;
  int e = threadIdx.x >> 5;
  float acc = 0.f;
  for (int i = lane; i < 2 * T_TOK; i += 32)
    if (sel[i] == e) acc += rw[i];
  for (int o = 16; o > 0; o >>= 1) acc += __shfl_xor(acc, o, 32);
  if (lane == 0) s_e[e] = acc;
  if (threadIdx.x == 0) {
    int r = 0;
    for (int k = 0; k < E_EXP; ++k) { offs[k] = r; r += counts[k]; }
  }
}

// --------------------- fused gate/up (SwiGLU) GEMM --------------------------
// Aout[row, n] = silu(X@W1^T) * (X@W3^T). Block tile 64x128, 8 waves of
// 32x32; B tiles via TDM, A tile via per-lane async loads; double-buffered.
__global__ void __launch_bounds__(256)
swiglu_gemm(const unsigned short* __restrict__ X,   // [T, K] bf16
            const unsigned short* __restrict__ W1,  // [Z, N, K] bf16
            const unsigned short* __restrict__ W3,  // [Z, N, K] bf16
            unsigned short* __restrict__ Aout,      // [rows, N] bf16
            const int* __restrict__ rowlist,        // [Z, T] or null
            const int* __restrict__ counts,         // [Z] or null
            const int* __restrict__ offs,           // [Z] or null
            int Mfixed, int N, int K) {
  const int e = blockIdx.z;
  const int M = counts ? counts[e] : Mfixed;
  const int m0 = blockIdx.y * 64;
  if (m0 >= M) return;
  const int n0 = blockIdx.x * 128;
  const unsigned short* w1 = W1 + (size_t)e * N * K;
  const unsigned short* w3 = W3 + (size_t)e * N * K;
  const int* lst = rowlist ? rowlist + (size_t)e * T_TOK : nullptr;
  const int rowoff = offs ? offs[e] : 0;

  __shared__ unsigned short lA[2][64 * KT];
  __shared__ unsigned short lB1[2][128 * KT];
  __shared__ unsigned short lB3[2][128 * KT];

  const int tid = threadIdx.x;
  const int lane = tid & 31;
  const int wv = tid >> 5;
  const int wm = wv >> 2;   // 0..1
  const int wn = wv & 3;    // 0..3
  const int half = lane >> 4;
  const int lrow = lane & 15;

  const int arow = tid >> 2;            // 0..63
  const int aseg = (tid & 3) * 8;       // 0,8,16,24
  int lr = m0 + arow;
  int grow = lst ? lst[(lr < M) ? lr : (M - 1)] : lr;
  const unsigned short* aptr = X + (size_t)grow * K + aseg;

  const unsigned short* b1base = w1 + (size_t)n0 * K;
  const unsigned short* b3base = w3 + (size_t)n0 * K;

  v8f accg[2][2] = {};
  v8f accu[2][2] = {};

  auto issue = [&](int buf, int k0) {
    async_ld_b128(&lA[buf][arow * KT + aseg], aptr + k0);
    if (wv == 0) {
      tdm_load_2d(&lB1[buf][0], b1base + k0, K, N, 128, KT);
      tdm_load_2d(&lB3[buf][0], b3base + k0, K, N, 128, KT);
    }
  };
  auto settle = [&]() {
    if (wv == 0) wait_tensor0();
    wait_async0();
    __syncthreads();
  };

  issue(0, 0);
  settle();

  const int nk = K / KT;
  for (int ki = 0; ki < nk; ++ki) {
    const int buf = ki & 1;
    if (ki + 1 < nk) issue(buf ^ 1, (ki + 1) * KT);

    FragU a[2], b1[2], b3[2];
    for (int mt = 0; mt < 2; ++mt) {
      int r = (wm * 32 + mt * 16 + lrow) * KT + half * 8;
      a[mt].q[0] = *(const uint4*)&lA[buf][r];
      a[mt].q[1] = *(const uint4*)&lA[buf][r + 16];
    }
    for (int nt = 0; nt < 2; ++nt) {
      int r = (wn * 32 + nt * 16 + lrow) * KT + half * 8;
      b1[nt].q[0] = *(const uint4*)&lB1[buf][r];
      b1[nt].q[1] = *(const uint4*)&lB1[buf][r + 16];
      b3[nt].q[0] = *(const uint4*)&lB3[buf][r];
      b3[nt].q[1] = *(const uint4*)&lB3[buf][r + 16];
    }
    for (int mt = 0; mt < 2; ++mt)
      for (int nt = 0; nt < 2; ++nt) {
        accg[mt][nt] = __builtin_amdgcn_wmma_f32_16x16x32_bf16(
            false, a[mt].v, false, b1[nt].v, (short)0, accg[mt][nt], false, false);
        accu[mt][nt] = __builtin_amdgcn_wmma_f32_16x16x32_bf16(
            false, a[mt].v, false, b3[nt].v, (short)0, accu[mt][nt], false, false);
      }
    settle();
  }

  for (int mt = 0; mt < 2; ++mt)
    for (int nt = 0; nt < 2; ++nt) {
      int col = n0 + wn * 32 + nt * 16 + lrow;
      for (int r = 0; r < 8; ++r) {
        int lm = m0 + wm * 32 + mt * 16 + r + half * 8;
        if (lm < M) {
          float g = accg[mt][nt][r];
          float u = accu[mt][nt][r];
          float aa = (g / (1.f + expf(-g))) * u;
          Aout[(size_t)(rowoff + lm) * N + col] = f2bf(aa);
        }
      }
    }
}

// ------------------------- down projection GEMM -----------------------------
__global__ void __launch_bounds__(256)
down_gemm(const unsigned short* __restrict__ Abuf,  // rows at rowoff, stride K
          const unsigned short* __restrict__ W,     // [Z, N, K] with wstride
          float* __restrict__ C,                    // rows at rowoff, stride N
          const int* __restrict__ counts,
          const int* __restrict__ offs,
          int Mfixed, int N, int K, size_t wstride) {
  const int e = blockIdx.z;
  const int M = counts ? counts[e] : Mfixed;
  const int m0 = blockIdx.y * 64;
  if (m0 >= M) return;
  const int n0 = blockIdx.x * 128;
  const unsigned short* w = W + (size_t)e * wstride;
  const int rowoff = offs ? offs[e] : 0;

  __shared__ unsigned short lA[2][64 * KT];
  __shared__ unsigned short lB[2][128 * KT];

  const int tid = threadIdx.x;
  const int lane = tid & 31;
  const int wv = tid >> 5;
  const int wm = wv >> 2;
  const int wn = wv & 3;
  const int half = lane >> 4;
  const int lrow = lane & 15;

  const int arow = tid >> 2;
  const int aseg = (tid & 3) * 8;
  int lr = m0 + arow;
  int clr = (lr < M) ? lr : (M - 1);
  const unsigned short* aptr = Abuf + (size_t)(rowoff + clr) * K + aseg;

  const unsigned short* bbase = w + (size_t)n0 * K;

  v8f acc[2][2] = {};

  auto issue = [&](int buf, int k0) {
    async_ld_b128(&lA[buf][arow * KT + aseg], aptr + k0);
    if (wv == 0)
      tdm_load_2d(&lB[buf][0], bbase + k0, K, N, 128, KT);
  };
  auto settle = [&]() {
    if (wv == 0) wait_tensor0();
    wait_async0();
    __syncthreads();
  };

  issue(0, 0);
  settle();

  const int nk = K / KT;
  for (int ki = 0; ki < nk; ++ki) {
    const int buf = ki & 1;
    if (ki + 1 < nk) issue(buf ^ 1, (ki + 1) * KT);

    FragU a[2], b[2];
    for (int mt = 0; mt < 2; ++mt) {
      int r = (wm * 32 + mt * 16 + lrow) * KT + half * 8;
      a[mt].q[0] = *(const uint4*)&lA[buf][r];
      a[mt].q[1] = *(const uint4*)&lA[buf][r + 16];
    }
    for (int nt = 0; nt < 2; ++nt) {
      int r = (wn * 32 + nt * 16 + lrow) * KT + half * 8;
      b[nt].q[0] = *(const uint4*)&lB[buf][r];
      b[nt].q[1] = *(const uint4*)&lB[buf][r + 16];
    }
    for (int mt = 0; mt < 2; ++mt)
      for (int nt = 0; nt < 2; ++nt)
        acc[mt][nt] = __builtin_amdgcn_wmma_f32_16x16x32_bf16(
            false, a[mt].v, false, b[nt].v, (short)0, acc[mt][nt], false, false);
    settle();
  }

  for (int mt = 0; mt < 2; ++mt)
    for (int nt = 0; nt < 2; ++nt) {
      int col = n0 + wn * 32 + nt * 16 + lrow;
      for (int r = 0; r < 8; ++r) {
        int lm = m0 + wm * 32 + mt * 16 + r + half * 8;
        if (lm < M)
          C[(size_t)(rowoff + lm) * N + col] = acc[mt][nt][r];
      }
    }
}

// ------------------------------- combine ------------------------------------
__global__ void __launch_bounds__(256)
combine_kernel(float* __restrict__ out, const float* __restrict__ ybuf,
               const int* __restrict__ sel, const int* __restrict__ posmap,
               const int* __restrict__ offs, const float* __restrict__ s_e) {
  int t = blockIdx.x;
  int e0 = sel[2 * t], e1 = sel[2 * t + 1];
  float s0 = s_e[e0], s1 = s_e[e1];
  size_t g0 = (size_t)(offs[e0] + posmap[2 * t]) * D_DIM;
  size_t g1 = (size_t)(offs[e1] + posmap[2 * t + 1]) * D_DIM;
  size_t o = (size_t)t * D_DIM;
  for (int d = threadIdx.x; d < D_DIM; d += blockDim.x)
    out[o + d] += s0 * ybuf[g0 + d] + s1 * ybuf[g1 + d];
}

// ---------------------------------------------------------------------------
extern "C" void kernel_launch(void* const* d_in, const int* in_sizes, int n_in,
                              void* d_out, int out_size, void* d_ws,
                              size_t ws_size, hipStream_t stream) {
  (void)in_sizes; (void)n_in; (void)out_size; (void)ws_size;
  const float* x   = (const float*)d_in[0];
  const float* gw  = (const float*)d_in[1];
  const float* w1  = (const float*)d_in[2];
  const float* w3  = (const float*)d_in[3];
  const float* w2  = (const float*)d_in[4];
  const float* sw1 = (const float*)d_in[5];
  const float* sw3 = (const float*)d_in[6];
  const float* sw2 = (const float*)d_in[7];
  float* out = (float*)d_out;

  char* p = (char*)d_ws;
  auto take = [&](size_t bytes) -> void* {
    void* r = (void*)p;
    p += (bytes + 255) & ~(size_t)255;
    return r;
  };
  const size_t nX  = (size_t)T_TOK * D_DIM;
  const size_t nW  = (size_t)E_EXP * H_DIM * D_DIM;
  const size_t nSW = (size_t)SH_DIM * D_DIM;
  unsigned short* xb   = (unsigned short*)take(nX * 2);
  unsigned short* w1b  = (unsigned short*)take(nW * 2);
  unsigned short* w3b  = (unsigned short*)take(nW * 2);
  unsigned short* w2b  = (unsigned short*)take(nW * 2);
  unsigned short* sw1b = (unsigned short*)take(nSW * 2);
  unsigned short* sw3b = (unsigned short*)take(nSW * 2);
  unsigned short* sw2b = (unsigned short*)take(nSW * 2);
  unsigned short* abuf = (unsigned short*)take((size_t)T_TOK * SH_DIM * 2); // == 2T*H
  float* ybuf  = (float*)take((size_t)2 * T_TOK * D_DIM * 4);
  int*   sel   = (int*)take((size_t)2 * T_TOK * 4);
  float* rw    = (float*)take((size_t)2 * T_TOK * 4);
  int*   pos   = (int*)take((size_t)2 * T_TOK * 4);
  int*   cnt   = (int*)take(E_EXP * 4);
  int*   list  = (int*)take((size_t)E_EXP * T_TOK * 4);
  float* s_e   = (float*)take(E_EXP * 4);
  int*   offs  = (int*)take(E_EXP * 4);

  (void)hipMemsetAsync(cnt, 0, E_EXP * sizeof(int), stream);

  // fp32 -> bf16 conversions
  cvt_f32_bf16<<<4096, 256, 0, stream>>>(x,   xb,   nX);
  cvt_f32_bf16<<<4096, 256, 0, stream>>>(w1,  w1b,  nW);
  cvt_f32_bf16<<<4096, 256, 0, stream>>>(w3,  w3b,  nW);
  cvt_f32_bf16<<<4096, 256, 0, stream>>>(w2,  w2b,  nW);
  cvt_f32_bf16<<<2048, 256, 0, stream>>>(sw1, sw1b, nSW);
  cvt_f32_bf16<<<2048, 256, 0, stream>>>(sw3, sw3b, nSW);
  cvt_f32_bf16<<<2048, 256, 0, stream>>>(sw2, sw2b, nSW);

  // routing
  gate_kernel<<<T_TOK / 8, 256, 0, stream>>>(x, gw, sel, rw, cnt, list, pos);
  se_off_kernel<<<1, 256, 0, stream>>>(sel, rw, cnt, s_e, offs);

  // shared expert: SwiGLU (N=SH over all T tokens), then down proj -> out
  swiglu_gemm<<<dim3(SH_DIM / 128, T_TOK / 64, 1), 256, 0, stream>>>(
      xb, sw1b, sw3b, abuf, nullptr, nullptr, nullptr, T_TOK, SH_DIM, D_DIM);
  down_gemm<<<dim3(D_DIM / 128, T_TOK / 64, 1), 256, 0, stream>>>(
      abuf, sw2b, out, nullptr, nullptr, T_TOK, D_DIM, SH_DIM, 0);

  // routed experts: gathered rows per expert (counts from gating)
  swiglu_gemm<<<dim3(H_DIM / 128, T_TOK / 64, E_EXP), 256, 0, stream>>>(
      xb, w1b, w3b, abuf, list, cnt, offs, 0, H_DIM, D_DIM);
  down_gemm<<<dim3(D_DIM / 128, T_TOK / 64, E_EXP), 256, 0, stream>>>(
      abuf, w2b, ybuf, cnt, offs, 0, D_DIM, H_DIM, (size_t)D_DIM * H_DIM);

  // out += s_e0 * y0 + s_e1 * y1 (fixed order, deterministic)
  combine_kernel<<<T_TOK, 256, 0, stream>>>(out, ybuf, sel, pos, offs, s_e);
}
